// FeedForwardNet_79877801771243
// MI455X (gfx1250) — matrix operand order repaired and checked by reading, not compiled, not run on hardware
//
#include <hip/hip_runtime.h>
#include <hip/hip_bf16.h>

// ---- structural constants (match reference) --------------------------------
#define NUM_IN 4096
#define LVLS   8
#define H      32768
#define KFAN   32
#define O      256
#define KO     64
#define BATCH  128
#define NODES  (NUM_IN + LVLS * H + O)   // 266496
#define SCALE  4.9f

typedef __attribute__((ext_vector_type(16))) _Float16 v16h;
typedef __attribute__((ext_vector_type(8)))  _Float16 v8h;
typedef __attribute__((ext_vector_type(4)))  _Float16 v4h;
typedef __attribute__((ext_vector_type(8)))  float    v8f;

// sigmoid(SCALE*x) with native HW ops: v_exp_f32 is exp2, v_rcp_f32 is rcp.
// exp(-SCALE*x) = exp2(x * (-SCALE*log2(e))) -> single fused constant mul.
__device__ __forceinline__ float fast_sigmoid_scaled(float pre) {
  const float kNegScaleLog2e = -SCALE * 1.44269504088896340736f;
  float e = __builtin_amdgcn_exp2f(pre * kNegScaleLog2e);
  return __builtin_amdgcn_rcpf(1.0f + e);
}

// ---------------------------------------------------------------------------
// Kernel 1: build the transposed fp16 activation buffer  buf[node][batch].
// buf[n][b] = x[b][n] for inputs, 1.0 for all not-yet-computed nodes.
// 68 MB total -> resident in the 192 MB L2 for the rest of the run.
// ---------------------------------------------------------------------------
__global__ void __launch_bounds__(256) init_buf_kernel(const float* __restrict__ x,
                                                       _Float16* __restrict__ buf) {
  int i = (blockIdx.x * 256 + threadIdx.x) * 4;  // 4 elems/thread, exact grid
  int n = i >> 7;                                // node (4 elems share one n)
  int b = i & 127;                               // batch
  v4h v;
#pragma unroll
  for (int j = 0; j < 4; ++j)
    v[j] = (_Float16)((n < NUM_IN) ? x[(b + j) * NUM_IN + n] : 1.0f);
  *(v4h*)(buf + i) = v;
}

// ---------------------------------------------------------------------------
// Kernel 2: one topological level. One wave32 per hidden unit.
//   A[m,k] = buf[idx[h][k]][b0+m]   (gathered activations, fp16, 16 batch x 32)
//   B[k,n] = w[h][k]                (replicated across N columns)
//   D[m,n] = dot(h, batch b0+m)     (replicated across n)
// A layout:              lane m   -> row m, K = 0..15
//                        lane m+16-> row m, K = 16..31
// B layout:              lanes 0-15 -> K=0..15, lanes 16-31 -> K=16..31
// ---------------------------------------------------------------------------
__global__ void __launch_bounds__(256) hidden_level_kernel(
    const float* __restrict__ w,     // w_hidden + l*H*K  [H][32]
    const int*   __restrict__ idx,   // idx_hidden + l*H*K
    _Float16* __restrict__ buf,
    int node_base)                   // NUM_IN + l*H
{
  const int lane = threadIdx.x & 31;
  const int wave = __builtin_amdgcn_readfirstlane((int)(threadIdx.x >> 5));
  const int h    = blockIdx.x * 8 + wave;        // hidden unit, 0..H-1 (exact)
  const int b    = lane & 15;                    // batch-in-tile = row M
  const bool hi  = lane >= 16;

  const int*   idxrow = idx + h * KFAN;
  const float* wrow   = w   + h * KFAN;

  // B operand: weights, replicated across the 16 N columns.
  v16h bmat;
#pragma unroll
  for (int e = 0; e < 16; ++e)
    bmat[e] = (_Float16)wrow[(hi ? 16 : 0) + e];

  // The 16 fan-in indices this lane's A slots need.
  int nidx[16];
#pragma unroll
  for (int j = 0; j < 16; ++j) {
    int k = j + (hi ? 16 : 0);
    nidx[j] = idxrow[k];
  }

  _Float16* dstrow = buf + node_base * BATCH + h * BATCH;

  for (int bt = 0; bt < 8; ++bt) {               // 8 tiles of 16 batch = 128
    const int b0 = bt * 16;
    v16h a;
#pragma unroll
    for (int j = 0; j < 16; ++j)                 // coalesced 2B x 16-lane gathers
      a[j] = buf[nidx[j] * BATCH + b0 + b];

    v8f c = {};
    c = __builtin_amdgcn_wmma_f32_16x16x32_f16(false, a, false, bmat,
                                               (short)0, c, false, false);

    // D rows replicated over N: lane 0 holds batches 0..7 in c[0..7],
    // lane 16 holds batches 8..15. sigmoid -> fp16 -> one 16B store each.
    v8h s;
#pragma unroll
    for (int m = 0; m < 8; ++m)
      s[m] = (_Float16)fast_sigmoid_scaled(c[m]);

    if (lane == 0)  *(v8h*)(dstrow + b0)     = s;
    if (lane == 16) *(v8h*)(dstrow + b0 + 8) = s;
  }
}

// ---------------------------------------------------------------------------
// Kernel 3: output layer. KO=64 -> two chained K=32 WMMAs per tile.
// One wave per output unit; result written fp32 to out[b][o].
// ---------------------------------------------------------------------------
__global__ void __launch_bounds__(256) out_layer_kernel(
    const float* __restrict__ wo,    // [O][64]
    const int*   __restrict__ idxo,  // [O][64]
    const _Float16* __restrict__ buf,
    float* __restrict__ out)         // [BATCH][O]
{
  const int lane = threadIdx.x & 31;
  const int wave = __builtin_amdgcn_readfirstlane((int)(threadIdx.x >> 5));
  const int o    = blockIdx.x * 8 + wave;        // 0..255 (exact)
  const int b    = lane & 15;
  const bool hi  = lane >= 16;

  const int*   idxrow = idxo + o * KO;
  const float* wrow   = wo   + o * KO;

  v16h bmat0, bmat1;
#pragma unroll
  for (int e = 0; e < 16; ++e) {
    bmat0[e] = (_Float16)wrow[(hi ? 16 : 0) + e];
    bmat1[e] = (_Float16)wrow[32 + (hi ? 16 : 0) + e];
  }

  int nidx[32];
#pragma unroll
  for (int j = 0; j < 16; ++j) {
    int k = j + (hi ? 16 : 0);
    nidx[j]      = idxrow[k];          // k in 0..31
    nidx[16 + j] = idxrow[32 + k];     // k in 32..63
  }

  for (int bt = 0; bt < 8; ++bt) {
    const int b0 = bt * 16;
    v16h a0, a1;
#pragma unroll
    for (int j = 0; j < 16; ++j) {
      a0[j] = buf[nidx[j]      * BATCH + b0 + b];
      a1[j] = buf[nidx[16 + j] * BATCH + b0 + b];
    }
    v8f c = {};
    c = __builtin_amdgcn_wmma_f32_16x16x32_f16(false, a0, false, bmat0,
                                               (short)0, c, false, false);
    c = __builtin_amdgcn_wmma_f32_16x16x32_f16(false, a1, false, bmat1,
                                               (short)0, c, false, false);

    if (lane == 0) {
#pragma unroll
      for (int m = 0; m < 8; ++m)
        out[(b0 + m) * O + o] = fast_sigmoid_scaled(c[m]);
    }
    if (lane == 16) {
#pragma unroll
      for (int m = 0; m < 8; ++m)
        out[(b0 + 8 + m) * O + o] = fast_sigmoid_scaled(c[m]);
    }
  }
}

// ---------------------------------------------------------------------------
extern "C" void kernel_launch(void* const* d_in, const int* in_sizes, int n_in,
                              void* d_out, int out_size, void* d_ws, size_t ws_size,
                              hipStream_t stream) {
  const float* x        = (const float*)d_in[0];   // [B, NUM_IN]
  const float* w_hidden = (const float*)d_in[1];   // [L, H, K]
  const float* w_out    = (const float*)d_in[2];   // [O, KO]
  const int*   idx_hid  = (const int*)  d_in[3];   // [L, H, K]
  const int*   idx_out  = (const int*)  d_in[4];   // [O, KO]
  float*       out      = (float*)d_out;           // [B, O]

  _Float16* buf = (_Float16*)d_ws;                 // [NODES][BATCH], ~68 MB

  // 1) transpose inputs + init ones (4 elems / thread, exact grid)
  init_buf_kernel<<<(NODES * BATCH) / (256 * 4), 256, 0, stream>>>(x, buf);

  // 2) 8 sequential topological levels (inter-level dep -> separate launches)
  for (int l = 0; l < LVLS; ++l) {
    hidden_level_kernel<<<H / 8, 256, 0, stream>>>(
        w_hidden + (size_t)l * H * KFAN,
        idx_hid  + (size_t)l * H * KFAN,
        buf,
        NUM_IN + l * H);
  }

  // 3) output layer
  out_layer_kernel<<<O / 8, 256, 0, stream>>>(w_out, idx_out, buf, out);
}
